// VisualTokenSelection_22119081574783
// MI455X (gfx1250) — compile-verified, matrix operand order checked by reading, not run on hardware
//
#include <hip/hip_runtime.h>
#include <hip/hip_bf16.h>
#include <math.h>

// Problem constants (from reference)
#define MAXF   12
#define TOPKN  49
#define NSAMP  500
#define SIGMA  0.05f
#define EPSLN  1e-5f
#define BPRIME 96      // B * MAX_FRAMES
#define NTOK   196
#define NTOKP  208     // padded to 13 full 16-row WMMA tiles
#define DDIM   512
#define HDIM   256

typedef __attribute__((ext_vector_type(16))) __bf16       v16bf;
typedef __attribute__((ext_vector_type(8)))  float        v8f;
typedef __attribute__((ext_vector_type(4)))  float        v4f;
typedef __attribute__((ext_vector_type(4)))  unsigned int v4u;

union F16x { v4f q[4]; float f[16]; };     // 4 x b128 worth of f32
union B16x { v4u q[2]; v16bf v; };         // 2 x b128 worth of bf16

// ---------- branch-free transcendentals (v_exp_f32 / v_rcp_f32 TRANS ops) ----------
#define LOG2E 1.4426950408889634f
__device__ inline float exp_fast(float x) { return __builtin_amdgcn_exp2f(x * LOG2E); }
__device__ inline float erf_fast(float x) {            // Abramowitz-Stegun 7.1.26, branchless
  float ax = fabsf(x);
  float t  = __builtin_amdgcn_rcpf(fmaf(0.3275911f, ax, 1.0f));
  float p  = t * fmaf(t, fmaf(t, fmaf(t, fmaf(t, 1.061405429f, -1.453152027f),
                                      1.421413741f), -0.284496736f), 0.254829592f);
  float r  = fmaf(-p, exp_fast(-ax * ax), 1.0f);
  return copysignf(r, x);
}
__device__ inline float geluf(float x) { return 0.5f * x * (1.0f + erf_fast(x * 0.70710678118f)); }
__device__ inline float tanh_fast(float x) {           // branchless via exp2
  float xc = fminf(fmaxf(x, -15.0f), 15.0f);
  float e  = exp_fast(2.0f * xc);
  return (e - 1.0f) * __builtin_amdgcn_rcpf(e + 1.0f);
}

// ---------- wave helpers ----------
__device__ inline float wredf(float v) {
  #pragma unroll
  for (int o = 16; o > 0; o >>= 1) v += __shfl_xor(v, o, 32);
  return v;
}
__device__ inline int wredi(int v) {
  #pragma unroll
  for (int o = 16; o > 0; o >>= 1) v += __shfl_xor(v, o, 32);
  return v;
}
__device__ inline int wexcl(int v) {   // wave32 exclusive prefix sum
  int lane = threadIdx.x & 31;
  int s = v;
  #pragma unroll
  for (int o = 1; o < 32; o <<= 1) {
    int t = __shfl_up(s, o, 32);
    if (lane >= o) s += t;
  }
  return s - v;
}
__device__ inline unsigned int h32(unsigned int x) {
  x ^= x >> 16; x *= 0x7feb352du; x ^= x >> 15; x *= 0x846ca68bu; x ^= x >> 16;
  return x;
}
__device__ inline float gaussf(unsigned int seed) {  // deterministic counter-based normal
  unsigned int r1 = h32(seed * 2u + 1u);
  unsigned int r2 = h32(seed * 2u + 0x9e3779b9u);
  float u1 = (float)(r1 | 1u) * 2.3283064365386963e-10f;   // (0,1]
  float u2 = (float)r2 * 2.3283064365386963e-10f;
  return sqrtf(-2.0f * __logf(u1)) * __cosf(6.28318530718f * u2);
}
__device__ inline unsigned int fkey(float f) {       // order-preserving float->uint
  unsigned int u = __builtin_bit_cast(unsigned int, f);
  return (u & 0x80000000u) ? ~u : (u | 0x80000000u);
}

// ---------- kernel 0: LayerNorm stats for every x token (one wave / token) ----------
__global__ __launch_bounds__(256) void k_lnstats(const float* __restrict__ x,
                                                 float* __restrict__ meanx,
                                                 float* __restrict__ rstdx, int ntok) {
  int wv = (int)((blockIdx.x * 256u + threadIdx.x) >> 5);
  int lane = threadIdx.x & 31;
  if (wv >= ntok) return;
  const float* r = x + (size_t)wv * DDIM;
  float s = 0.f, s2 = 0.f;
  for (int i = lane; i < DDIM; i += 32) { float v = r[i]; s += v; s2 += v * v; }
  s = wredf(s); s2 = wredf(s2);
  if (lane == 0) {
    float m = s * (1.0f / DDIM);
    float var = s2 * (1.0f / DDIM) - m * m;
    meanx[wv] = m;
    rstdx[wv] = rsqrtf(var + EPSLN);
  }
}

// ---------- kernel 1: transpose+convert weights to bf16 (B-fragment friendly) ----------
__global__ __launch_bounds__(256) void k_wprep(const float* __restrict__ Win,
                                               const float* __restrict__ Wout1,
                                               __bf16* __restrict__ WbTin,
                                               __bf16* __restrict__ WbTo1) {
  int i = blockIdx.x * 256 + threadIdx.x;
  if (i < DDIM * HDIM) {                         // WbTin[n][k] = Win[k][n]
    int n = i / DDIM, k = i % DDIM;
    WbTin[i] = (__bf16)Win[(size_t)k * HDIM + n];
  } else {
    int j = i - DDIM * HDIM;
    if (j < HDIM * HDIM) {                       // WbTo1[n][k] = Wout1[k][n], k<256 (top half)
      int n = j / HDIM, k = j % HDIM;
      WbTo1[j] = (__bf16)Wout1[(size_t)k * HDIM + n];
    }
  }
}

// ---------- kernel 2: guidance vectors: gvec = gelu(LN(guid)@W_gd) @ W_out1[256:,:] ----------
__global__ __launch_bounds__(256) void k_guid(const float* __restrict__ gframe,
                                              const float* __restrict__ gsent,
                                              const float* __restrict__ lg,
                                              const float* __restrict__ lb,
                                              const float* __restrict__ Wgd,
                                              const float* __restrict__ Wout1,
                                              float* __restrict__ gvec) {
  int bp = blockIdx.x;
  int tid = threadIdx.x, wid = tid >> 5, lane = tid & 31;
  __shared__ float lnq[2][DDIM];
  __shared__ float loc[2][HDIM];
  __shared__ float rb[8];
  const float* srcs[2] = { gframe + (size_t)bp * DDIM,
                           gsent + (size_t)(bp / MAXF) * DDIM };
  for (int t = 0; t < 2; ++t) {
    const float* sp = srcs[t];
    float s = 0.f, s2 = 0.f;
    for (int i = tid; i < DDIM; i += 256) { float v = sp[i]; s += v; s2 += v * v; }
    s = wredf(s);
    if (lane == 0) rb[wid] = s;
    __syncthreads();
    float S = 0.f; for (int q = 0; q < 8; ++q) S += rb[q];
    __syncthreads();
    s2 = wredf(s2);
    if (lane == 0) rb[wid] = s2;
    __syncthreads();
    float S2 = 0.f; for (int q = 0; q < 8; ++q) S2 += rb[q];
    __syncthreads();
    float mean = S * (1.0f / DDIM);
    float rstd = rsqrtf(S2 * (1.0f / DDIM) - mean * mean + EPSLN);
    for (int i = tid; i < DDIM; i += 256) lnq[t][i] = (sp[i] - mean) * rstd * lg[i] + lb[i];
    __syncthreads();
  }
  for (int t = 0; t < 2; ++t) {
    float a = 0.f;
    for (int k = 0; k < DDIM; ++k) a += lnq[t][k] * Wgd[(size_t)k * HDIM + tid];
    loc[t][tid] = geluf(a);
  }
  __syncthreads();
  for (int t = 0; t < 2; ++t) {
    float a = 0.f;
    for (int k = 0; k < HDIM; ++k) a += loc[t][k] * Wout1[(size_t)(HDIM + k) * HDIM + tid];
    gvec[(size_t)bp * 2 * HDIM + t * HDIM + tid] = a;
  }
}

// ---------- kernel 3: GEMM1 via WMMA: local = gelu(LN(x) @ W_in), stored bf16, 208-row padded ----------
// One 16x16 output tile per wave; all 13x16 = 208 rows stored unconditionally (no predication).
__global__ __launch_bounds__(256) void k_gemm1(const float* __restrict__ x,
                                               const float* __restrict__ meanx,
                                               const float* __restrict__ rstdx,
                                               const float* __restrict__ lng,
                                               const float* __restrict__ lnb,
                                               const __bf16* __restrict__ WbTin,
                                               __bf16* __restrict__ localb) {
  int bp = blockIdx.x;
  int wid = threadIdx.x >> 5, lane = threadIdx.x & 31;
  int tile = blockIdx.y * 8 + wid;            // 0..207
  int mt = tile % 13, nt = tile / 13;
  int half = lane >> 4, lr = lane & 15;
  int m = mt * 16 + lr;
  int mc = (m < NTOK) ? m : (NTOK - 1);       // clamp source row, don't predicate loads
  const float* xr = x + ((size_t)bp * NTOK + mc) * DDIM;
  float mean = meanx[bp * NTOK + mc];
  float rstd = rstdx[bp * NTOK + mc];
  int n = nt * 16 + lr;
  const __bf16* br = WbTin + (size_t)n * DDIM;
  v8f c = {0.f, 0.f, 0.f, 0.f, 0.f, 0.f, 0.f, 0.f};
  for (int kt = 0; kt < DDIM / 32; ++kt) {
    int kb = kt * 32 + half * 8;              // lane's two 8-elem runs: [kb, kb+8) and [kb+16, kb+24)
    F16x xa, ga, ba;
    xa.q[0] = *(const v4f*)(xr + kb);      xa.q[1] = *(const v4f*)(xr + kb + 4);
    xa.q[2] = *(const v4f*)(xr + kb + 16); xa.q[3] = *(const v4f*)(xr + kb + 20);
    ga.q[0] = *(const v4f*)(lng + kb);      ga.q[1] = *(const v4f*)(lng + kb + 4);
    ga.q[2] = *(const v4f*)(lng + kb + 16); ga.q[3] = *(const v4f*)(lng + kb + 20);
    ba.q[0] = *(const v4f*)(lnb + kb);      ba.q[1] = *(const v4f*)(lnb + kb + 4);
    ba.q[2] = *(const v4f*)(lnb + kb + 16); ba.q[3] = *(const v4f*)(lnb + kb + 20);
    v16bf a;
    #pragma unroll
    for (int e = 0; e < 16; ++e)
      a[e] = (__bf16)((xa.f[e] - mean) * rstd * ga.f[e] + ba.f[e]);
    B16x bb;
    bb.q[0] = *(const v4u*)(br + kb);
    bb.q[1] = *(const v4u*)(br + kb + 16);
    c = __builtin_amdgcn_wmma_f32_16x16x32_bf16(false, a, false, bb.v, (short)0, c, false, false);
  }
  // C: row = r + 8*half, col = lr; store all 208 padded rows unconditionally
  __bf16* orow = localb + ((size_t)bp * NTOKP + mt * 16 + half * 8) * HDIM + nt * 16 + lr;
  #pragma unroll
  for (int r = 0; r < 8; ++r)
    orow[(size_t)r * HDIM] = (__bf16)geluf(c[r]);
}

// ---------- kernel 4: GEMM2 via WMMA + fused score epilogue ----------
// score = tanh(gelu(local@W1top + gvecF)@w2) + tanh(gelu(local@W1top + gvecS)@w2)
// grid(96, 13) x 8 waves; wave handles n-tiles {wid, wid+8}; per-row dot reduced in LDS.
__global__ __launch_bounds__(256) void k_score(const __bf16* __restrict__ localb,
                                               const __bf16* __restrict__ WbTo1,
                                               const float* __restrict__ gvec,
                                               const float* __restrict__ w2,
                                               float* __restrict__ scores) {
  int bp = blockIdx.x, mt = blockIdx.y;
  int wid = threadIdx.x >> 5, lane = threadIdx.x & 31;
  int half = lane >> 4, lr = lane & 15;
  __shared__ float acc[16][2];
  if (threadIdx.x < 32) acc[threadIdx.x >> 1][threadIdx.x & 1] = 0.f;
  __syncthreads();
  const __bf16* ar = localb + ((size_t)bp * NTOKP + mt * 16 + lr) * HDIM;  // padded: no clamp
  const float* gF = gvec + (size_t)bp * 2 * HDIM;
  const float* gS = gF + HDIM;
  for (int pass = 0; pass < 2; ++pass) {
    int nt = wid + pass * 8;
    int n = nt * 16 + lr;
    const __bf16* br = WbTo1 + (size_t)n * HDIM;
    v8f c = {0.f, 0.f, 0.f, 0.f, 0.f, 0.f, 0.f, 0.f};
    for (int kt = 0; kt < HDIM / 32; ++kt) {
      int kb = kt * 32 + half * 8;
      B16x aa, bb;
      aa.q[0] = *(const v4u*)(ar + kb);
      aa.q[1] = *(const v4u*)(ar + kb + 16);
      bb.q[0] = *(const v4u*)(br + kb);
      bb.q[1] = *(const v4u*)(br + kb + 16);
      c = __builtin_amdgcn_wmma_f32_16x16x32_bf16(false, aa.v, false, bb.v, (short)0, c, false, false);
    }
    float w2v = w2[n];
    float gf = gF[n], gs = gS[n];
    #pragma unroll
    for (int r = 0; r < 8; ++r) {
      int row = r + half * 8;
      float cv = c[r];
      atomicAdd(&acc[row][0], geluf(cv + gf) * w2v);
      atomicAdd(&acc[row][1], geluf(cv + gs) * w2v);
    }
  }
  __syncthreads();
  if (threadIdx.x < 16) {
    int mo = mt * 16 + threadIdx.x;
    if (mo < NTOK)
      scores[(size_t)bp * NTOK + mo] = tanh_fast(acc[threadIdx.x][0]) + tanh_fast(acc[threadIdx.x][1]);
  }
}

// ---------- kernel 5: perturbed top-k soft indicators (one wave / sample) ----------
__global__ __launch_bounds__(256) void k_topk(const float* __restrict__ scores,
                                              float* __restrict__ ind) {
  int bp = blockIdx.x;
  int wid = threadIdx.x >> 5, lane = threadIdx.x & 31;
  __shared__ float sc[NTOK];
  for (int i = threadIdx.x; i < NTOK; i += 256) sc[i] = scores[(size_t)bp * NTOK + i];
  __syncthreads();
  int s = blockIdx.y * 8 + wid;
  if (s >= NSAMP) return;
  unsigned int keys[7];
  int dbase = lane * 7;                         // lane owns contiguous indices
  #pragma unroll
  for (int j = 0; j < 7; ++j) {
    int d = dbase + j;
    if (d < NTOK) {
      unsigned int seed = ((unsigned)bp * NSAMP + (unsigned)s) * NTOK + (unsigned)d;
      keys[j] = fkey(sc[d] + SIGMA * gaussf(seed));
    } else keys[j] = 0u;
  }
  // 49th-largest key via bitwise binary search with wave reductions
  unsigned int thr = 0u;
  for (int bit = 31; bit >= 0; --bit) {
    unsigned int t = thr | (1u << bit);
    int cl = 0;
    #pragma unroll
    for (int j = 0; j < 7; ++j) cl += (keys[j] >= t) ? 1 : 0;
    if (wredi(cl) >= TOPKN) thr = t;
  }
  int cg = 0, ce = 0;
  #pragma unroll
  for (int j = 0; j < 7; ++j) { cg += (keys[j] > thr); ce += (keys[j] == thr); }
  int sg = wexcl(cg), se = wexcl(ce);
  int cg_tot = __shfl(sg + cg, 31, 32);
  int need = TOPKN - cg_tot;                    // ties admitted in index order
  #pragma unroll
  for (int j = 0; j < 7; ++j) {
    int d = dbase + j;
    if (d >= NTOK) break;
    unsigned int k = keys[j];
    if (k > thr) {
      int rank = sg + ((se < need) ? se : need);
      if (rank < TOPKN)
        atomicAdd(&ind[((size_t)bp * TOPKN + rank) * NTOK + d], 1.0f / NSAMP);
      sg++;
    } else if (k == thr) {
      if (se < need) {
        int rank = sg + se;
        if (rank < TOPKN)
          atomicAdd(&ind[((size_t)bp * TOPKN + rank) * NTOK + d], 1.0f / NSAMP);
      }
      se++;
    }
  }
}

// ---------- kernel 6: selected = ind @ x  (f32, zero-skip; grid(96,4)) ----------
__global__ __launch_bounds__(256) void k_select(const float* __restrict__ ind,
                                                const float* __restrict__ x,
                                                float* __restrict__ out) {
  int bp = blockIdx.x, kb = blockIdx.y;
  __shared__ float sInd[13 * NTOK];
  for (int i = threadIdx.x; i < 13 * NTOK; i += 256) {
    int kk = i / NTOK, l = i % NTOK, k = kb * 13 + kk;
    sInd[i] = (k < TOPKN) ? ind[((size_t)bp * TOPKN + k) * NTOK + l] : 0.f;
  }
  __syncthreads();
  const float* xb = x + (size_t)bp * NTOK * DDIM;
  int d0 = threadIdx.x, d1 = threadIdx.x + 256;
  for (int kk = 0; kk < 13; ++kk) {
    int k = kb * 13 + kk;
    if (k >= TOPKN) break;
    float a0 = 0.f, a1 = 0.f;
    for (int l = 0; l < NTOK; ++l) {
      float w = sInd[kk * NTOK + l];
      if (w != 0.f) {                            // wave-uniform skip of zero columns
        const float* xr = xb + (size_t)l * DDIM;
        a0 += w * xr[d0];
        a1 += w * xr[d1];
      }
    }
    float* o = out + ((size_t)bp * TOPKN + k) * DDIM;
    o[d0] = a0; o[d1] = a1;
  }
}

// ---------- launcher ----------
extern "C" void kernel_launch(void* const* d_in, const int* in_sizes, int n_in,
                              void* d_out, int out_size, void* d_ws, size_t ws_size,
                              hipStream_t stream) {
  (void)in_sizes; (void)n_in; (void)out_size; (void)ws_size;
  const float* x       = (const float*)d_in[0];   // (8, 2352, 512)
  const float* gframe  = (const float*)d_in[1];   // (8, 12, 512)
  const float* gsent   = (const float*)d_in[2];   // (8, 1, 512)
  const float* ln_in_g = (const float*)d_in[3];
  const float* ln_in_b = (const float*)d_in[4];
  const float* W_in    = (const float*)d_in[5];   // (512, 256)
  const float* ln_gd_g = (const float*)d_in[6];
  const float* ln_gd_b = (const float*)d_in[7];
  const float* W_gd    = (const float*)d_in[8];   // (512, 256)
  const float* W_out1  = (const float*)d_in[9];   // (512, 256)
  const float* W_out2  = (const float*)d_in[10];  // (256, 1)
  float* out = (float*)d_out;                     // (8, 12, 49, 512)

  // workspace layout (~14.8 MB total, all sub-buffers 16B-aligned)
  char* w = (char*)d_ws;
  float* meanx = (float*)w;
  float* rstdx = meanx + BPRIME * NTOK;
  float* gvec  = rstdx + BPRIME * NTOK;                         // (96, 2, 256)
  __bf16* WbTin  = (__bf16*)(gvec + BPRIME * 2 * HDIM);
  __bf16* WbTo1  = WbTin + (size_t)HDIM * DDIM;
  __bf16* localb = WbTo1 + (size_t)HDIM * HDIM;                 // (96, 208, 256) bf16, row-padded
  float* scores = (float*)(localb + (size_t)BPRIME * NTOKP * HDIM);
  float* ind    = scores + BPRIME * NTOK;                       // (96, 49, 196)

  // indicators are accumulated with atomics -> must be zeroed every call
  hipMemsetAsync(ind, 0, (size_t)BPRIME * TOPKN * NTOK * sizeof(float), stream);

  k_lnstats<<<dim3((BPRIME * NTOK + 7) / 8), 256, 0, stream>>>(x, meanx, rstdx, BPRIME * NTOK);
  k_wprep<<<dim3((DDIM * HDIM + HDIM * HDIM) / 256), 256, 0, stream>>>(W_in, W_out1, WbTin, WbTo1);
  k_guid<<<dim3(BPRIME), 256, 0, stream>>>(gframe, gsent, ln_gd_g, ln_gd_b, W_gd, W_out1, gvec);
  k_gemm1<<<dim3(BPRIME, 26), 256, 0, stream>>>(x, meanx, rstdx, ln_in_g, ln_in_b, WbTin, localb);
  k_score<<<dim3(BPRIME, 13), 256, 0, stream>>>(localb, WbTo1, gvec, W_out2, scores);
  k_topk<<<dim3(BPRIME, (NSAMP + 7) / 8), 256, 0, stream>>>(scores, ind);
  k_select<<<dim3(BPRIME, 4), 256, 0, stream>>>(ind, x, out);
}
